// Topology_Encoder_4810363371521
// MI455X (gfx1250) — matrix operand order repaired and checked by reading, not compiled, not run on hardware
//
#include <hip/hip_runtime.h>
#include <hip/hip_bf16.h>

// ---------------------------------------------------------------------------
// Topology encoder for MI455X (gfx1250, wave32, WMMA).
// N=4096, K=128, ALPHA=0.5, N_ITERS=2, TOPK=2048, ROW_TOPK=128
// ---------------------------------------------------------------------------

static constexpr int NN   = 4096;
static constexpr int KC   = 128;     // candidate count / embd dim
static constexpr int KT   = 32;      // GEMM K step (WMMA bf16 K)
static constexpr float EPSF = 1e-10f;

typedef __bf16 bf16;
typedef __attribute__((ext_vector_type(16))) __bf16 v16bf;
typedef __attribute__((ext_vector_type(8)))  float  v8f;

union FragU { v16bf v; uint4 q[2]; };

// ---------------------------------------------------------------------------
// deg[i] = sum_j adj[i][j]
__global__ __launch_bounds__(256)
void te_rowsum_kernel(const float* __restrict__ adj, float* __restrict__ deg) {
    __shared__ float red[256];
    int row = blockIdx.x;
    float s = 0.f;
    for (int j = threadIdx.x; j < NN; j += 256) s += adj[(size_t)row * NN + j];
    red[threadIdx.x] = s; __syncthreads();
    for (int off = 128; off > 0; off >>= 1) {
        if (threadIdx.x < off) red[threadIdx.x] += red[threadIdx.x + off];
        __syncthreads();
    }
    if (threadIdx.x == 0) deg[row] = red[0];
}

// P_bf16[i][j] = bf16( adj[i][j] / (deg[i] + eps) )
__global__ __launch_bounds__(256)
void te_makeP_kernel(const float* __restrict__ adj, const float* __restrict__ deg,
                     bf16* __restrict__ P) {
    int idx = blockIdx.x * 256 + threadIdx.x;           // < NN*NN
    int row = idx >> 12;
    P[idx] = (bf16)(adj[idx] / (deg[row] + EPSF));
}

// ---------------------------------------------------------------------------
// C = 0.5 * (A_bf16 @ B_bf16) + adj/(deg+eps)   (the "M = (1-a)PM + P" step)
// Block tile 128x256, 8 waves (2x4), wave tile 64x64 -> 16 WMMA / K-step.
// As: row-major 128x32. Bs: COLUMN-major [col*32+k] so each lane's B fragment
// is two 16B runs, k 8h..8h+7 and 16+8h..16+8h+7 (2x ds_load_b128).
__global__ __launch_bounds__(256)
void te_gemm_pm_kernel(const bf16* __restrict__ A, const bf16* __restrict__ B,
                       const float* __restrict__ adj, const float* __restrict__ deg,
                       float* __restrict__ Cf, bf16* __restrict__ Cb) {
    __shared__ bf16 As[128 * KT];    // 8 KB, row-major
    __shared__ bf16 Bs[256 * KT];    // 16 KB, column-major: Bs[c*32 + k]
    const int tid  = threadIdx.x;
    const int wave = tid >> 5;
    const int lane = tid & 31;
    const int wm   = wave >> 2;      // 0..1 : 64-row band
    const int wn   = wave & 3;       // 0..3 : 64-col band
    const int lm   = lane & 15;
    const int lh   = lane >> 4;
    const int brow = blockIdx.y * 128;
    const int bcol = blockIdx.x * 256;

    v8f acc[4][4] = {};

    for (int k0 = 0; k0 < NN; k0 += KT) {
        // stage A tile (128x32 row-major): 4 x (4 bf16 = 8B) per thread
        #pragma unroll
        for (int r = 0; r < 4; r++) {
            int idx = (tid + r * 256) * 4;
            int row = idx >> 5, col = idx & 31;
            *(uint2*)&As[idx] =
                *(const uint2*)&A[(size_t)(brow + row) * NN + k0 + col];
        }
        // stage B tile (32x256 in memory -> col-major in LDS):
        // coalesced 8B global reads, 4 b16 scatter stores each
        #pragma unroll
        for (int r = 0; r < 8; r++) {
            int idx = (tid + r * 256) * 4;      // < 8192
            int k = idx >> 8, c = idx & 255;
            bf16 tmp[4];
            *(uint2*)tmp = *(const uint2*)&B[(size_t)(k0 + k) * NN + bcol + c];
            #pragma unroll
            for (int t = 0; t < 4; t++) Bs[(c + t) * KT + k] = tmp[t];
        }
        if (k0 + KT < NN)  // keep next A rows warm (global_prefetch_b8)
            __builtin_prefetch(&A[(size_t)(brow + (tid >> 1)) * NN + k0 + KT], 0, 1);
        __syncthreads();

        // A fragments: contiguous 16B runs (ISA 16-bit A 16x32 layout)
        FragU af[4];
        #pragma unroll
        for (int mi = 0; mi < 4; mi++) {
            int arow = wm * 64 + mi * 16 + lm;
            af[mi].q[0] = *(const uint4*)&As[arow * KT + lh * 8];
            af[mi].q[1] = *(const uint4*)&As[arow * KT + 16 + lh * 8];
        }
        // B fragments: two 16B runs per lane, same k order as A (col-major Bs)
        FragU bf_[4];
        #pragma unroll
        for (int ni = 0; ni < 4; ni++) {
            int bc = wn * 64 + ni * 16 + lm;
            bf_[ni].q[0] = *(const uint4*)&Bs[bc * KT + lh * 8];
            bf_[ni].q[1] = *(const uint4*)&Bs[bc * KT + 16 + lh * 8];
        }
        #pragma unroll
        for (int mi = 0; mi < 4; mi++)
            #pragma unroll
            for (int ni = 0; ni < 4; ni++)
                acc[mi][ni] = __builtin_amdgcn_wmma_f32_16x16x32_bf16(
                    false, af[mi].v, false, bf_[ni].v, (short)0, acc[mi][ni],
                    false, false);
        __syncthreads();
    }

    // epilogue: M = 0.5*acc + adj/(deg+eps)   (C layout: VGPR v -> M=v+8*lh, N=lm)
    #pragma unroll
    for (int mi = 0; mi < 4; mi++) {
        #pragma unroll
        for (int ni = 0; ni < 4; ni++) {
            #pragma unroll
            for (int v = 0; v < 8; v++) {
                int m = wm * 64 + mi * 16 + v + 8 * lh;
                int c = wn * 64 + ni * 16 + lm;
                size_t g = (size_t)(brow + m) * NN + (bcol + c);
                float p = adj[g] / (deg[brow + m] + EPSF);
                float val = 0.5f * acc[mi][ni][v] + p;
                if (Cf) Cf[g] = val;
                if (Cb) Cb[g] = (bf16)val;
            }
        }
    }
}

// ---------------------------------------------------------------------------
// sim = e @ e^T  (e: NNxKC bf16, f32 accumulate), diag forced to 0
// Block 128x128, wave tile 32x64. Bs is col-major; since E is row-major in
// (n,k), staging is contiguous on BOTH the global and LDS side.
__global__ __launch_bounds__(256)
void te_sim_gemm_kernel(const bf16* __restrict__ E, float* __restrict__ S) {
    __shared__ bf16 As[128 * KT];
    __shared__ bf16 Bs[128 * KT];    // col-major: Bs[c*32 + k]
    const int tid  = threadIdx.x;
    const int wave = tid >> 5;
    const int lane = tid & 31;
    const int wm   = wave >> 1;      // 0..3
    const int wn   = wave & 1;       // 0..1
    const int lm   = lane & 15;
    const int lh   = lane >> 4;
    const int brow = blockIdx.y * 128;
    const int bcol = blockIdx.x * 128;

    v8f acc[2][4] = {};

    for (int k0 = 0; k0 < KC; k0 += KT) {
        #pragma unroll
        for (int r = 0; r < 4; r++) {
            int idx = (tid + r * 256) * 4;
            int row = idx >> 5, col = idx & 31;
            *(uint2*)&As[idx] =
                *(const uint2*)&E[(size_t)(brow + row) * KC + k0 + col];
        }
        #pragma unroll
        for (int r = 0; r < 4; r++) {
            int idx = (tid + r * 256) * 4;
            int c = idx >> 5, k = idx & 31;
            *(uint2*)&Bs[c * KT + k] =
                *(const uint2*)&E[(size_t)(bcol + c) * KC + k0 + k];
        }
        __syncthreads();

        FragU af[2];
        #pragma unroll
        for (int mi = 0; mi < 2; mi++) {
            int arow = wm * 32 + mi * 16 + lm;
            af[mi].q[0] = *(const uint4*)&As[arow * KT + lh * 8];
            af[mi].q[1] = *(const uint4*)&As[arow * KT + 16 + lh * 8];
        }
        FragU bf_[4];
        #pragma unroll
        for (int ni = 0; ni < 4; ni++) {
            int bc = wn * 64 + ni * 16 + lm;
            bf_[ni].q[0] = *(const uint4*)&Bs[bc * KT + lh * 8];
            bf_[ni].q[1] = *(const uint4*)&Bs[bc * KT + 16 + lh * 8];
        }
        #pragma unroll
        for (int mi = 0; mi < 2; mi++)
            #pragma unroll
            for (int ni = 0; ni < 4; ni++)
                acc[mi][ni] = __builtin_amdgcn_wmma_f32_16x16x32_bf16(
                    false, af[mi].v, false, bf_[ni].v, (short)0, acc[mi][ni],
                    false, false);
        __syncthreads();
    }

    #pragma unroll
    for (int mi = 0; mi < 2; mi++)
        #pragma unroll
        for (int ni = 0; ni < 4; ni++)
            #pragma unroll
            for (int v = 0; v < 8; v++) {
                int gi = brow + wm * 32 + mi * 16 + v + 8 * lh;
                int gj = bcol + wn * 64 + ni * 16 + lm;
                float val = (gi == gj) ? 0.f : acc[mi][ni][v];
                S[(size_t)gi * NN + gj] = val;
            }
}

// ---------------------------------------------------------------------------
// column sums of M (coalesced: one thread per column)
__global__ __launch_bounds__(256)
void te_colsum_kernel(const float* __restrict__ M, float* __restrict__ cs) {
    int j = blockIdx.x * 256 + threadIdx.x;
    float s = 0.f;
    for (int i = 0; i < NN; i++) s += M[(size_t)i * NN + j];
    cs[j] = s;
}

// top-128 columns by colsum (descending, stable tie -> lower index)
__global__ __launch_bounds__(256)
void te_cand_topk_kernel(const float* __restrict__ cs, int* __restrict__ cand) {
    __shared__ float vals[NN];
    __shared__ float rv[256];
    __shared__ int   ri[256];
    for (int j = threadIdx.x; j < NN; j += 256) vals[j] = cs[j];
    __syncthreads();
    for (int t = 0; t < KC; t++) {
        float bv = -1e30f; int bi = 0;
        for (int j = threadIdx.x; j < NN; j += 256) {
            float v = vals[j];
            if (v > bv) { bv = v; bi = j; }
        }
        rv[threadIdx.x] = bv; ri[threadIdx.x] = bi; __syncthreads();
        for (int off = 128; off > 0; off >>= 1) {
            if (threadIdx.x < off) {
                float v2 = rv[threadIdx.x + off]; int i2 = ri[threadIdx.x + off];
                if (v2 > rv[threadIdx.x] ||
                    (v2 == rv[threadIdx.x] && i2 < ri[threadIdx.x])) {
                    rv[threadIdx.x] = v2; ri[threadIdx.x] = i2;
                }
            }
            __syncthreads();
        }
        if (threadIdx.x == 0) { cand[t] = ri[0]; vals[ri[0]] = -1e30f; }
        __syncthreads();
    }
}

__global__ __launch_bounds__(256)
void te_gather_mk_kernel(const float* __restrict__ M, const int* __restrict__ cand,
                         float* __restrict__ Mk) {
    int idx = blockIdx.x * 256 + threadIdx.x;   // NN*KC
    int i = idx >> 7, k = idx & 127;
    Mk[idx] = M[(size_t)i * NN + cand[k]];
}

__global__ __launch_bounds__(256)
void te_mk_rowfac_kernel(const float* __restrict__ Mk, float* __restrict__ colf) {
    int i = blockIdx.x * 256 + threadIdx.x;
    float s = 0.f;
    for (int k = 0; k < KC; k++) s += Mk[(size_t)i * KC + k];
    colf[i] = 1.f / (sqrtf(s) + EPSF);
}

__global__ __launch_bounds__(128)
void te_mk_colfac_kernel(const float* __restrict__ Mk, float* __restrict__ rowf) {
    int k = threadIdx.x;
    float s = 0.f;
    for (int i = 0; i < NN; i++) s += Mk[(size_t)i * KC + k];
    rowf[k] = 1.f / (sqrtf(s) + EPSF);
}

// center_idx[i] = argmax_k colf[i]*Mk[i,k]*rowf[k]  (first-max, like jnp.argmax)
__global__ __launch_bounds__(256)
void te_center_kernel(const float* __restrict__ Mk, const float* __restrict__ colf,
                      const float* __restrict__ rowf, int* __restrict__ cidx) {
    int i = blockIdx.x * 256 + threadIdx.x;
    float cf = colf[i];
    float bv = -1e30f; int bk = 0;
    for (int k = 0; k < KC; k++) {
        float p = cf * Mk[(size_t)i * KC + k] * rowf[k];
        if (p > bv) { bv = p; bk = k; }
    }
    cidx[i] = bk;
}

// embd[i,k] = sum_j adj[i,j]*(cidx[j]==k)  via LDS float atomics; also e = l2norm(row)
__global__ __launch_bounds__(256)
void te_embd_kernel(const float* __restrict__ adj, const int* __restrict__ cidx,
                    float* __restrict__ embd_out, bf16* __restrict__ Eb) {
    __shared__ float h[KC];
    __shared__ float red[256];
    int i = blockIdx.x;
    if (threadIdx.x < KC) h[threadIdx.x] = 0.f;
    __syncthreads();
    for (int j = threadIdx.x; j < NN; j += 256) {
        float a = adj[(size_t)i * NN + j];
        if (a != 0.f) atomicAdd(&h[cidx[j]], a);     // ds_add_f32
    }
    __syncthreads();
    float s = 0.f;
    if (threadIdx.x < KC) { float v = h[threadIdx.x]; s = v * v; }
    red[threadIdx.x] = s; __syncthreads();
    for (int off = 128; off > 0; off >>= 1) {
        if (threadIdx.x < off) red[threadIdx.x] += red[threadIdx.x + off];
        __syncthreads();
    }
    float nrm = sqrtf(red[0]);
    float inv = 1.f / (nrm == 0.f ? EPSF : nrm);
    if (threadIdx.x < KC) {
        float v = h[threadIdx.x];
        embd_out[(size_t)i * KC + threadIdx.x] = v;
        Eb[(size_t)i * KC + threadIdx.x] = (bf16)(v * inv);
    }
}

// per-row top-128 of sim -> writes full row_t row into st (rest zero)
__global__ __launch_bounds__(256)
void te_rowtopk_kernel(const float* __restrict__ S, float* __restrict__ st) {
    __shared__ float vals[NN];   // 16 KB
    __shared__ float outv[NN];   // 16 KB  (320KB LDS/WGP makes this cheap)
    __shared__ float rv[256];
    __shared__ int   ri[256];
    int i = blockIdx.x;
    for (int j = threadIdx.x; j < NN; j += 256) {
        vals[j] = S[(size_t)i * NN + j];
        outv[j] = 0.f;
    }
    __syncthreads();
    for (int t = 0; t < 128; t++) {
        float bv = -1e30f; int bi = 0;
        for (int j = threadIdx.x; j < NN; j += 256) {
            float v = vals[j];
            if (v > bv) { bv = v; bi = j; }
        }
        rv[threadIdx.x] = bv; ri[threadIdx.x] = bi; __syncthreads();
        for (int off = 128; off > 0; off >>= 1) {
            if (threadIdx.x < off) {
                float v2 = rv[threadIdx.x + off]; int i2 = ri[threadIdx.x + off];
                if (v2 > rv[threadIdx.x] ||
                    (v2 == rv[threadIdx.x] && i2 < ri[threadIdx.x])) {
                    rv[threadIdx.x] = v2; ri[threadIdx.x] = i2;
                }
            }
            __syncthreads();
        }
        if (threadIdx.x == 0) { outv[ri[0]] = rv[0]; vals[ri[0]] = -1e30f; }
        __syncthreads();
    }
    for (int j = threadIdx.x; j < NN; j += 256) st[(size_t)i * NN + j] = outv[j];
}

// ---- global top-2048 threshold via 2-level radix histogram on float bits ----
__global__ __launch_bounds__(256)
void te_zero_u32_kernel(unsigned int* __restrict__ p, int n) {
    int i = blockIdx.x * 256 + threadIdx.x;
    if (i < n) p[i] = 0u;
}

__global__ __launch_bounds__(256)
void te_hist_kernel(const float* __restrict__ S, unsigned int* __restrict__ hist) {
    int idx = blockIdx.x * 256 + threadIdx.x;
    unsigned int bits = __float_as_uint(S[idx]);   // sims are >= 0 -> monotonic
    atomicAdd(&hist[bits >> 16], 1u);
}

__global__ void te_thresh1_kernel(const unsigned int* __restrict__ hist,
                                  unsigned int* __restrict__ meta) {
    if (threadIdx.x != 0) return;
    unsigned long long cum = 0; int bin = 0;
    for (int b = 65535; b >= 0; b--) { cum += hist[b]; if (cum >= 2048ull) { bin = b; break; } }
    unsigned long long above = 0;
    for (int b = 65535; b > bin; b--) above += hist[b];
    meta[0] = (unsigned)bin;
    meta[1] = (unsigned)(2048ull - above);   // still needed within threshold bin
}

__global__ __launch_bounds__(256)
void te_hist2_kernel(const float* __restrict__ S, const unsigned int* __restrict__ meta,
                     unsigned int* __restrict__ hist2) {
    int idx = blockIdx.x * 256 + threadIdx.x;
    unsigned int bits = __float_as_uint(S[idx]);
    if ((bits >> 16) == meta[0]) atomicAdd(&hist2[bits & 0xFFFFu], 1u);
}

__global__ void te_thresh2_kernel(const unsigned int* __restrict__ hist2,
                                  unsigned int* __restrict__ meta) {
    if (threadIdx.x != 0) return;
    unsigned need = meta[1];
    if (need == 0u) {
        meta[2] = (meta[0] >= 65535u) ? 0xFFFFFFFFu : ((meta[0] + 1u) << 16);
        return;
    }
    unsigned long long cum = 0; unsigned low = 0;
    for (int b = 65535; b >= 0; b--) { cum += hist2[b]; if (cum >= need) { low = (unsigned)b; break; } }
    meta[2] = (meta[0] << 16) | low;         // include all values with bits >= meta[2]
}

__global__ __launch_bounds__(256)
void te_addtopk_kernel(const float* __restrict__ S, const unsigned int* __restrict__ meta,
                       float* __restrict__ st) {
    int idx = blockIdx.x * 256 + threadIdx.x;
    float v = S[idx];
    unsigned int bits = __float_as_uint(v);
    if (bits >= meta[2] && v > 0.f) st[idx] += v;   // st = row_t + topk_t
}

// adjn, adj_kernel outputs (||adj row|| = sqrt(deg) exactly for 0/1 adj)
__global__ __launch_bounds__(256)
void te_output_kernel(const float* __restrict__ adj, const float* __restrict__ deg,
                      const float* __restrict__ st, float* __restrict__ out_adjn,
                      float* __restrict__ out_kern) {
    __shared__ float red[256];
    int i = blockIdx.x;
    float s = 0.f;
    for (int j = threadIdx.x; j < NN; j += 256) {
        float v = st[(size_t)i * NN + j]; s += v * v;
    }
    red[threadIdx.x] = s; __syncthreads();
    for (int off = 128; off > 0; off >>= 1) {
        if (threadIdx.x < off) red[threadIdx.x] += red[threadIdx.x + off];
        __syncthreads();
    }
    float sn = sqrtf(red[0]);
    float invs = 1.f / (sn == 0.f ? EPSF : sn);
    float an = sqrtf(deg[i]);
    float inva = 1.f / (an == 0.f ? EPSF : an);
    for (int j = threadIdx.x; j < NN; j += 256) {
        size_t g = (size_t)i * NN + j;
        float a = adj[g] * inva;
        float t = st[g] * invs;
        out_adjn[g] = a;
        float d = a - t;
        out_kern[g] = 0.5f * (a + t) + d * d;
    }
}

// ---------------------------------------------------------------------------
extern "C" void kernel_launch(void* const* d_in, const int* in_sizes, int n_in,
                              void* d_out, int out_size, void* d_ws, size_t ws_size,
                              hipStream_t stream) {
    (void)in_sizes; (void)n_in; (void)out_size; (void)ws_size;
    const float* adj = (const float*)d_in[0];   // central/labeled masks unused by forward

    float* out        = (float*)d_out;
    float* out_embd   = out;                                    // NN x KC
    float* out_adjn   = out + (size_t)NN * KC;                  // NN x NN
    float* out_kern   = out_adjn + (size_t)NN * NN;             // NN x NN

    // workspace carve-up (256B aligned)
    char* ws = (char*)d_ws;
    size_t off = 0;
    auto carve = [&](size_t bytes) { char* p = ws + off; off = (off + bytes + 255) & ~(size_t)255; return p; };
    float*        deg   = (float*)       carve((size_t)NN * 4);
    bf16*         Pb    = (bf16*)        carve((size_t)NN * NN * 2);
    bf16*         M1b   = (bf16*)        carve((size_t)NN * NN * 2);
    float*        M2    = (float*)       carve((size_t)NN * NN * 4);
    float*        cs    = (float*)       carve((size_t)NN * 4);
    int*          cand  = (int*)         carve((size_t)KC * 4);
    float*        Mk    = (float*)       carve((size_t)NN * KC * 4);
    float*        colf  = (float*)       carve((size_t)NN * 4);
    float*        rowf  = (float*)       carve((size_t)KC * 4);
    int*          cidx  = (int*)         carve((size_t)NN * 4);
    bf16*         Eb    = (bf16*)        carve((size_t)NN * KC * 2);
    float*        S     = (float*)       carve((size_t)NN * NN * 4);
    float*        st    = (float*)       carve((size_t)NN * NN * 4);
    unsigned int* hist  = (unsigned int*)carve(65536 * 4);
    unsigned int* hist2 = (unsigned int*)carve(65536 * 4);
    unsigned int* meta  = (unsigned int*)carve(16 * 4);

    const int EW = (NN * NN) / 256;     // elementwise grid (65536)
    dim3 ggPM(NN / 256, NN / 128);      // 16 x 32  (block tile 128x256)
    dim3 ggS (NN / 128, NN / 128);      // 32 x 32  (block tile 128x128)

    te_rowsum_kernel  <<<NN, 256, 0, stream>>>(adj, deg);
    te_makeP_kernel   <<<EW, 256, 0, stream>>>(adj, deg, Pb);
    // M1 = 0.5*(P@P) + P  (bf16 only, feeds next GEMM)
    te_gemm_pm_kernel <<<ggPM, 256, 0, stream>>>(Pb, Pb, adj, deg, nullptr, M1b);
    // M2 = 0.5*(P@M1) + P (f32)
    te_gemm_pm_kernel <<<ggPM, 256, 0, stream>>>(Pb, M1b, adj, deg, M2, nullptr);

    te_colsum_kernel   <<<NN / 256, 256, 0, stream>>>(M2, cs);
    te_cand_topk_kernel<<<1, 256, 0, stream>>>(cs, cand);
    te_gather_mk_kernel<<<(NN * KC) / 256, 256, 0, stream>>>(M2, cand, Mk);
    te_mk_rowfac_kernel<<<NN / 256, 256, 0, stream>>>(Mk, colf);
    te_mk_colfac_kernel<<<1, 128, 0, stream>>>(Mk, rowf);
    te_center_kernel   <<<NN / 256, 256, 0, stream>>>(Mk, colf, rowf, cidx);
    te_embd_kernel     <<<NN, 256, 0, stream>>>(adj, cidx, out_embd, Eb);

    te_sim_gemm_kernel <<<ggS, 256, 0, stream>>>(Eb, S);

    te_rowtopk_kernel  <<<NN, 256, 0, stream>>>(S, st);

    te_zero_u32_kernel <<<(2 * 65536 + 16 + 255) / 256, 256, 0, stream>>>(hist, 2 * 65536 + 16);
    te_hist_kernel     <<<EW, 256, 0, stream>>>(S, hist);
    te_thresh1_kernel  <<<1, 64, 0, stream>>>(hist, meta);
    te_hist2_kernel    <<<EW, 256, 0, stream>>>(S, meta, hist2);
    te_thresh2_kernel  <<<1, 64, 0, stream>>>(hist2, meta);
    te_addtopk_kernel  <<<EW, 256, 0, stream>>>(S, meta, st);

    te_output_kernel   <<<NN, 256, 0, stream>>>(adj, deg, st, out_adjn, out_kern);
}
